// ReMMTASformer_52432960750134
// MI455X (gfx1250) — matrix operand order repaired and checked by reading, not compiled, not run on hardware
//
#include <hip/hip_runtime.h>
#include <hip/hip_bf16.h>
#include <math.h>

typedef __bf16 bf16;
typedef __attribute__((ext_vector_type(16))) __bf16 v16bf;
typedef __attribute__((ext_vector_type(8)))  __bf16 v8bf;
typedef __attribute__((ext_vector_type(8)))  float  v8f;

#define WMMA_BF16(A_, B_, C_) \
  __builtin_amdgcn_wmma_f32_16x16x32_bf16(false, (A_), false, (B_), (short)0, (C_), false, false)

// ---- problem dims ----
#define NB   2
#define NS   4096
#define ND   1024
#define NL   512
#define NST  64
#define NH   8
#define NT   640              // 2*NST + NL
#define NDH  4096
#define NBT  (NB * NT)        // 1280
#define NBS  (NB * NS)        // 8192

// =====================================================================
// WMMA fragment loaders (wave32; this file's contiguous k order, same for A and B)
// A 16x32 bf16: lanes 0-15 row M=lane, elems=K0..15 contiguous;
//               lanes 16-31 same rows, K16..31.
// B 32x16 bf16 (loaded from N-major "Bt[n][k]"): lanes 0-15 col N=lane,
//               elems=K0..15 contiguous; lanes 16-31 K16..31.
// C/D 16x16 f32: elem r -> row r + 8*(lane>>4), col lane&15.
// =====================================================================
__device__ __forceinline__ v16bf ld_a_frag(const bf16* base, int ld) {
  const int lane = threadIdx.x & 31;
  const bf16* p = base + (long)(lane & 15) * ld + ((lane >> 4) << 4);
  v8bf lo = *reinterpret_cast<const v8bf*>(p);
  v8bf hi = *reinterpret_cast<const v8bf*>(p + 8);
  v16bf r;
#pragma unroll
  for (int i = 0; i < 8; ++i) { r[i] = lo[i]; r[i + 8] = hi[i]; }
  return r;
}

__device__ __forceinline__ v16bf ld_b_frag(const bf16* base, int ld) {
  const int lane = threadIdx.x & 31;
  const bf16* p = base + (long)(lane & 15) * ld + ((lane >> 4) << 4);
  return *reinterpret_cast<const v16bf*>(p);
}

__device__ __forceinline__ float gelu_exact(float v) {
  return 0.5f * v * (1.0f + erff(v * 0.70710678118654752f));
}

// =====================================================================
// Generic bf16 GEMM, double-buffered LDS, one barrier per K-step.
// C[z] = A[z] (MxK row-major) * Bt[z%bmod]^T (Bt stored NxK).
// Block: 128 threads (4 waves, 2x2). Block tile 64 x (2*WN).
// Each wave computes 32 x WN  ->  2*(WN/16) WMMA per 32-wide K-step.
// All M,N,K used here are multiples of the tile sizes -> no edge guards.
// =====================================================================
template <int WN>   // wave N-tile: 64 (large GEMMs) or 32 (per-head GEMMs)
__global__ __launch_bounds__(128)
void gemm_bf16(const bf16* __restrict__ A, long sAz, int lda,
               const bf16* __restrict__ Bt, long sBz, int bmod, int ldb,
               float* __restrict__ Cf, bf16* __restrict__ Cb, long sCz, int ldc,
               const float* __restrict__ bias, int act_gelu,
               int M, int N, int K) {
  constexpr int BN = 2 * WN;           // block N tile (128 or 64)
  constexpr int BIT = BN / 64;         // B-tile rows loaded per thread (2 or 1)
  constexpr int NJ = WN / 16;          // N-fragments per wave (4 or 2)

  const int z = blockIdx.z;
  const bf16* Ab = A + (long)z * sAz;
  const bf16* Bb = Bt + (long)(z % bmod) * sBz;
  const long cbase = (long)z * sCz;

  const int m0 = blockIdx.y * 64;
  const int n0 = blockIdx.x * BN;

  __shared__ __attribute__((aligned(32))) bf16 As[2][64 * 32];
  __shared__ __attribute__((aligned(32))) bf16 Bs[2][BN * 32];

  const int tid  = threadIdx.x;
  const int lane = tid & 31;
  const int wave = tid >> 5;
  const int wm = (wave >> 1) * 32;
  const int wn = (wave & 1) * WN;
  const int hi = lane >> 4, ln = lane & 15;

  const int ld_row = tid >> 1;         // 0..63
  const int ld_off = (tid & 1) << 4;   // 0 or 16 elements (32B)

  v8f acc[2][NJ] = {};
  uint4 av[2], bv[BIT][2];

  auto gload = [&](int k0) {
    const uint4* ap =
        reinterpret_cast<const uint4*>(Ab + (long)(m0 + ld_row) * lda + k0 + ld_off);
    av[0] = ap[0]; av[1] = ap[1];
#pragma unroll
    for (int it = 0; it < BIT; ++it) {
      const uint4* bp = reinterpret_cast<const uint4*>(
          Bb + (long)(n0 + ld_row + 64 * it) * ldb + k0 + ld_off);
      bv[it][0] = bp[0]; bv[it][1] = bp[1];
    }
  };
  auto sstore = [&](int buf) {
    *reinterpret_cast<uint4*>(&As[buf][ld_row * 32 + ld_off])     = av[0];
    *reinterpret_cast<uint4*>(&As[buf][ld_row * 32 + ld_off + 8]) = av[1];
#pragma unroll
    for (int it = 0; it < BIT; ++it) {
      *reinterpret_cast<uint4*>(&Bs[buf][(ld_row + 64 * it) * 32 + ld_off])     = bv[it][0];
      *reinterpret_cast<uint4*>(&Bs[buf][(ld_row + 64 * it) * 32 + ld_off + 8]) = bv[it][1];
    }
  };

  const int nk = K / 32;
  gload(0);
  sstore(0);
  __syncthreads();

  for (int k = 0; k < nk; ++k) {
    const int buf = k & 1;
    if (k + 1 < nk) gload((k + 1) * 32);   // overlap next global fetch with math

    v16bf fa0 = ld_a_frag(&As[buf][(wm +  0) * 32], 32);
    v16bf fa1 = ld_a_frag(&As[buf][(wm + 16) * 32], 32);
#pragma unroll
    for (int j = 0; j < NJ; ++j) {
      v16bf fb = ld_b_frag(&Bs[buf][(wn + 16 * j) * 32], 32);
      acc[0][j] = WMMA_BF16(fa0, fb, acc[0][j]);
      acc[1][j] = WMMA_BF16(fa1, fb, acc[1][j]);
    }
    if (k + 1 < nk) {
      sstore(buf ^ 1);    // writes the buffer whose readers all retired last barrier
      __syncthreads();
    }
  }

#pragma unroll
  for (int i = 0; i < 2; ++i)
#pragma unroll
    for (int j = 0; j < NJ; ++j)
#pragma unroll
      for (int r = 0; r < 8; ++r) {
        float v = acc[i][j][r];
        const int gm = m0 + wm + 16 * i + r + 8 * hi;
        const int gn = n0 + wn + 16 * j + ln;
        if (bias) v += bias[gn];
        if (act_gelu) v = gelu_exact(v);
        const long off = cbase + (long)gm * ldc + gn;
        if (Cf) Cf[off] = v;
        else    Cb[off] = (bf16)v;
      }
}

// =====================================================================
// Flash attention, causal, T=640, head dim 64, scale=1/8.
// One wave per 16 query rows (single-wave workgroup: barrier ops -> S_NOP,
// only the needed s_wait_dscnt remains). 32-key tiles; QK^T and PV via
// WMMA; P converted D-layout -> A-layout through LDS; V pre-transposed.
// =====================================================================
__global__ __launch_bounds__(32)
void attn_flash(const bf16* __restrict__ q, long qsb, long qsh, int qst,
                const bf16* __restrict__ k, long ksb, long ksh, int kst,
                const bf16* __restrict__ vt,  // [(b*NH+h)][64][NT]
                bf16* __restrict__ o, long osb, long osh, int ost) {
  const int qt = blockIdx.x;       // query tile (16 rows)
  const int z  = blockIdx.y;       // b*NH + h
  const int b  = z / NH, h = z % NH;
  const int lane = threadIdx.x;
  const int hi = lane >> 4, ln = lane & 15;

  const bf16* qb  = q + (long)b * qsb + (long)h * qsh;
  const bf16* kb  = k + (long)b * ksb + (long)h * ksh;
  const bf16* vtb = vt + (long)z * 64 * NT;
  bf16*       ob  = o + (long)b * osb + (long)h * osh;

  __shared__ __attribute__((aligned(32))) bf16 Pb[16 * 32];

  const v16bf qa0 = ld_a_frag(qb + (long)(qt * 16) * qst +  0, qst);
  const v16bf qa1 = ld_a_frag(qb + (long)(qt * 16) * qst + 32, qst);

  float mrow[8], lrow[8];
  v8f O0 = {}, O1 = {}, O2 = {}, O3 = {};
#pragma unroll
  for (int r = 0; r < 8; ++r) { mrow[r] = -1e30f; lrow[r] = 0.f; }

  const int jmax = (qt * 16 + 15) / 32;
  for (int j = 0; j <= jmax; ++j) {
    const int key0 = j * 32;
    v8f S0 = {}, S1 = {};
    {
      v16bf k00 = ld_b_frag(kb + (long)(key0     ) * kst +  0, kst);
      v16bf k01 = ld_b_frag(kb + (long)(key0     ) * kst + 32, kst);
      v16bf k10 = ld_b_frag(kb + (long)(key0 + 16) * kst +  0, kst);
      v16bf k11 = ld_b_frag(kb + (long)(key0 + 16) * kst + 32, kst);
      S0 = WMMA_BF16(qa0, k00, S0);
      S0 = WMMA_BF16(qa1, k01, S0);
      S1 = WMMA_BF16(qa0, k10, S1);
      S1 = WMMA_BF16(qa1, k11, S1);
    }
    float p0[8], p1[8], alpha[8];
#pragma unroll
    for (int r = 0; r < 8; ++r) {
      const int row = qt * 16 + r + 8 * hi;
      float s0 = S0[r] * 0.125f;
      float s1 = S1[r] * 0.125f;
      if (key0 + ln      > row) s0 = -1e30f;
      if (key0 + 16 + ln > row) s1 = -1e30f;
      float mx = fmaxf(s0, s1);
#pragma unroll
      for (int off = 8; off > 0; off >>= 1) mx = fmaxf(mx, __shfl_xor(mx, off, 16));
      const float mn = fmaxf(mrow[r], mx);
      alpha[r] = __expf(mrow[r] - mn);
      const float e0 = __expf(s0 - mn);
      const float e1 = __expf(s1 - mn);
      float rs = e0 + e1;
#pragma unroll
      for (int off = 8; off > 0; off >>= 1) rs += __shfl_xor(rs, off, 16);
      lrow[r] = lrow[r] * alpha[r] + rs;
      mrow[r] = mn;
      p0[r] = e0; p1[r] = e1;
    }
#pragma unroll
    for (int r = 0; r < 8; ++r) {
      O0[r] *= alpha[r]; O1[r] *= alpha[r];
      O2[r] *= alpha[r]; O3[r] *= alpha[r];
      Pb[(r + 8 * hi) * 32 + ln]      = (bf16)p0[r];
      Pb[(r + 8 * hi) * 32 + 16 + ln] = (bf16)p1[r];
    }
    __syncthreads();
    const v16bf pa  = ld_a_frag(Pb, 32);
    const v16bf vb0 = ld_b_frag(vtb + (long)( 0) * NT + key0, NT);
    const v16bf vb1 = ld_b_frag(vtb + (long)(16) * NT + key0, NT);
    const v16bf vb2 = ld_b_frag(vtb + (long)(32) * NT + key0, NT);
    const v16bf vb3 = ld_b_frag(vtb + (long)(48) * NT + key0, NT);
    O0 = WMMA_BF16(pa, vb0, O0);
    O1 = WMMA_BF16(pa, vb1, O1);
    O2 = WMMA_BF16(pa, vb2, O2);
    O3 = WMMA_BF16(pa, vb3, O3);
    __syncthreads();
  }

#pragma unroll
  for (int r = 0; r < 8; ++r) {
    const int row = qt * 16 + r + 8 * hi;
    const float inv = 1.0f / lrow[r];
    bf16* op = ob + (long)row * ost + ln;
    op[ 0] = (bf16)(O0[r] * inv);
    op[16] = (bf16)(O1[r] * inv);
    op[32] = (bf16)(O2[r] * inv);
    op[48] = (bf16)(O3[r] * inv);
  }
}

// =====================================================================
// Elementwise helpers
// =====================================================================
// in f32 [z][R][C] -> out bf16 [z][C][R]  (weights -> N-major bf16)
__global__ void k_transpose_cvt(const float* __restrict__ in, bf16* __restrict__ out,
                                int R, int C, long total) {
  long i = (long)blockIdx.x * blockDim.x + threadIdx.x;
  if (i >= total) return;
  const int c = (int)(i % C);
  const long t = i / C;
  const int r = (int)(t % R);
  const long z = t / R;
  out[z * (long)R * C + (long)c * R + r] = (bf16)in[i];
}

__global__ void k_state_init(const float* __restrict__ sinit, float* __restrict__ state) {
  long i = (long)blockIdx.x * blockDim.x + threadIdx.x;
  if (i >= (long)NB * NST * ND) return;
  state[i] = sinit[i % ((long)NST * ND)];
}

// toks[b][t][d] = state | x segment | state   (bf16)
__global__ void k_build_toks(const float* __restrict__ x, const float* __restrict__ state,
                             bf16* __restrict__ toks, int seg) {
  long i = (long)blockIdx.x * blockDim.x + threadIdx.x;
  if (i >= (long)NB * NT * ND) return;
  const int d = (int)(i % ND);
  const int t = (int)((i / ND) % NT);
  const int b = (int)(i / ((long)ND * NT));
  float v;
  if (t < NST)            v = state[((long)b * NST + t) * ND + d];
  else if (t < NST + NL)  v = x[((long)b * NS + (long)seg * NL + (t - NST)) * ND + d];
  else                    v = state[((long)b * NST + (t - NST - NL)) * ND + d];
  toks[i] = (bf16)v;
}

// RoPE in-place on bf16 buffer; element (b,h,t,2p) at b*sb+h*sh+t*st+2p
__global__ void k_rope(bf16* __restrict__ buf, long sb, long sh, int st) {
  long i = (long)blockIdx.x * blockDim.x + threadIdx.x;
  if (i >= (long)NB * NH * NT * 32) return;
  const int p = (int)(i % 32);
  const int t = (int)((i / 32) % NT);
  const int h = (int)((i / (32L * NT)) % NH);
  const int b = (int)(i / (32L * NT * NH));
  const float inv = powf(10000.0f, -(float)(2 * p) / 64.0f);
  const float ang = (float)t * inv;
  const float c = cosf(ang), s = sinf(ang);
  const long base = (long)b * sb + (long)h * sh + (long)t * st + 2 * p;
  const float x1 = (float)buf[base];
  const float x2 = (float)buf[base + 1];
  buf[base]     = (bf16)(x1 * c - x2 * s);
  buf[base + 1] = (bf16)(x1 * s + x2 * c);
}

// v (strided) -> vt[(b*NH+h)][dv][t]
__global__ void k_transpose_v(const bf16* __restrict__ v, long sb, long sh, int st,
                              bf16* __restrict__ vt) {
  long i = (long)blockIdx.x * blockDim.x + threadIdx.x;
  if (i >= (long)NB * NH * NT * 64) return;
  const int dv = (int)(i % 64);
  const int t  = (int)((i / 64) % NT);
  const int h  = (int)((i / (64L * NT)) % NH);
  const int b  = (int)(i / (64L * NT * NH));
  vt[((long)(b * NH + h) * 64 + dv) * NT + t] =
      v[(long)b * sb + (long)h * sh + (long)t * st + dv];
}

// out f32 [b][NT][ND]: middle NL rows -> a_full, last NST rows -> state
__global__ void k_scatter(const float* __restrict__ out, float* __restrict__ a_full,
                          float* __restrict__ state, int seg) {
  long i = (long)blockIdx.x * blockDim.x + threadIdx.x;
  if (i >= (long)NB * NT * ND) return;
  const int d = (int)(i % ND);
  const int t = (int)((i / ND) % NT);
  const int b = (int)(i / ((long)ND * NT));
  const float v = out[i];
  if (t >= NST && t < NST + NL)
    a_full[((long)b * NS + (long)seg * NL + (t - NST)) * ND + d] = v;
  else if (t >= NST + NL)
    state[((long)b * NST + (t - NST - NL)) * ND + d] = v;
}

// =====================================================================
// LayerNorm over D=1024: out = (a+res - mu)*rstd*g + beta -> bf16 or f32
// =====================================================================
__device__ __forceinline__ float block_sum(float v, float* sh) {
#pragma unroll
  for (int o = 16; o > 0; o >>= 1) v += __shfl_xor(v, o, 32);
  const int w = threadIdx.x >> 5;
  if ((threadIdx.x & 31) == 0) sh[w] = v;
  __syncthreads();
  float t = 0.f;
#pragma unroll
  for (int i = 0; i < 8; ++i) t += sh[i];
  __syncthreads();
  return t;
}

__global__ __launch_bounds__(256)
void k_layernorm(const float* __restrict__ a, const float* __restrict__ res,
                 const float* __restrict__ g, const float* __restrict__ beta,
                 bf16* __restrict__ outb, float* __restrict__ outf) {
  __shared__ float sh[8];
  const long base = (long)blockIdx.x * ND;
  float loc[4];
  float sum = 0.f;
#pragma unroll
  for (int i = 0; i < 4; ++i) {
    const int d = threadIdx.x + 256 * i;
    const float v = a[base + d] + res[base + d];
    loc[i] = v; sum += v;
  }
  const float mu = block_sum(sum, sh) * (1.0f / ND);
  float vs = 0.f;
#pragma unroll
  for (int i = 0; i < 4; ++i) { const float dd = loc[i] - mu; vs += dd * dd; }
  const float var = block_sum(vs, sh) * (1.0f / ND);
  const float rstd = rsqrtf(var + 1e-5f);
#pragma unroll
  for (int i = 0; i < 4; ++i) {
    const int d = threadIdx.x + 256 * i;
    const float o = (loc[i] - mu) * rstd * g[d] + beta[d];
    if (outb) outb[base + d] = (bf16)o;
    else      outf[base + d] = o;
  }
}

// =====================================================================
// Host orchestration
// =====================================================================
static inline int ceil_div_i(long a, long b) { return (int)((a + b - 1) / b); }

extern "C" void kernel_launch(void* const* d_in, const int* in_sizes, int n_in,
                              void* d_out, int out_size, void* d_ws, size_t ws_size,
                              hipStream_t stream) {
  (void)in_sizes; (void)n_in; (void)out_size; (void)ws_size;
  const float* x      = (const float*)d_in[0];
  const float* sinit  = (const float*)d_in[1];
  const float* Wq0    = (const float*)d_in[2];
  const float* Wk0    = (const float*)d_in[3];
  const float* Wv0    = (const float*)d_in[4];
  const float* Wq1    = (const float*)d_in[5];
  const float* Wk1    = (const float*)d_in[6];
  const float* Wv1    = (const float*)d_in[7];
  const float* Wo     = (const float*)d_in[8];
  const float* bo     = (const float*)d_in[9];
  const float* g_attn = (const float*)d_in[10];
  const float* b_attn = (const float*)d_in[11];
  const float* W1     = (const float*)d_in[12];
  const float* b1     = (const float*)d_in[13];
  const float* W2     = (const float*)d_in[14];
  const float* b2     = (const float*)d_in[15];
  const float* g_mlp  = (const float*)d_in[16];
  const float* b_mlp  = (const float*)d_in[17];
  float* outp = (float*)d_out;

  char* wsb = (char*)d_ws;
  size_t off = 0;
  auto alloc = [&](size_t bytes) -> void* {
    void* p = wsb + off;
    off += (bytes + 255) & ~(size_t)255;
    return p;
  };

  // bf16 N-major weights
  bf16* wq0t = (bf16*)alloc((size_t)512 * 1024 * 2);
  bf16* wk0t = (bf16*)alloc((size_t)512 * 1024 * 2);
  bf16* wv0t = (bf16*)alloc((size_t)512 * 1024 * 2);
  bf16* wq1t = (bf16*)alloc((size_t)NH * 64 * 64 * 2);
  bf16* wk1t = (bf16*)alloc((size_t)NH * 64 * 64 * 2);
  bf16* wv1t = (bf16*)alloc((size_t)NH * 64 * 64 * 2);
  bf16* wot  = (bf16*)alloc((size_t)1024 * 512 * 2);
  bf16* w1t  = (bf16*)alloc((size_t)NDH * ND * 2);
  bf16* w2t  = (bf16*)alloc((size_t)ND * NDH * 2);
  // activations
  bf16* toks = (bf16*)alloc((size_t)NBT * ND * 2);
  bf16* q0   = (bf16*)alloc((size_t)NBT * 512 * 2);
  bf16* k0   = (bf16*)alloc((size_t)NBT * 512 * 2);
  bf16* v0   = (bf16*)alloc((size_t)NBT * 512 * 2);
  bf16* v0t  = (bf16*)alloc((size_t)NBT * 512 * 2);
  bf16* o0   = (bf16*)alloc((size_t)NBT * 512 * 2);
  bf16* q1   = (bf16*)alloc((size_t)NBT * 512 * 2);
  bf16* k1   = (bf16*)alloc((size_t)NBT * 512 * 2);
  bf16* v1   = (bf16*)alloc((size_t)NBT * 512 * 2);
  bf16* v1t  = (bf16*)alloc((size_t)NBT * 512 * 2);
  bf16* o1   = (bf16*)alloc((size_t)NBT * 512 * 2);
  float* outbuf = (float*)alloc((size_t)NBT * ND * 4);
  float* state  = (float*)alloc((size_t)NB * NST * ND * 4);
  float* a_full = (float*)alloc((size_t)NBS * ND * 4);
  bf16*  hbuf   = (bf16*)alloc((size_t)NBS * ND * 2);
  bf16*  m1     = (bf16*)alloc((size_t)NBS * NDH * 2);
  float* m2     = (float*)alloc((size_t)NBS * ND * 4);

  const int TPB = 256;
  auto ew = [&](long tot) { return dim3(ceil_div_i(tot, TPB)); };

  // --- weight convert+transpose ---
  k_transpose_cvt<<<ew((long)1024 * 512), TPB, 0, stream>>>(Wq0, wq0t, 1024, 512, (long)1024 * 512);
  k_transpose_cvt<<<ew((long)1024 * 512), TPB, 0, stream>>>(Wk0, wk0t, 1024, 512, (long)1024 * 512);
  k_transpose_cvt<<<ew((long)1024 * 512), TPB, 0, stream>>>(Wv0, wv0t, 1024, 512, (long)1024 * 512);
  k_transpose_cvt<<<ew((long)NH * 64 * 64), TPB, 0, stream>>>(Wq1, wq1t, 64, 64, (long)NH * 64 * 64);
  k_transpose_cvt<<<ew((long)NH * 64 * 64), TPB, 0, stream>>>(Wk1, wk1t, 64, 64, (long)NH * 64 * 64);
  k_transpose_cvt<<<ew((long)NH * 64 * 64), TPB, 0, stream>>>(Wv1, wv1t, 64, 64, (long)NH * 64 * 64);
  k_transpose_cvt<<<ew((long)512 * 1024), TPB, 0, stream>>>(Wo, wot, 512, 1024, (long)512 * 1024);
  k_transpose_cvt<<<ew((long)ND * NDH), TPB, 0, stream>>>(W1, w1t, ND, NDH, (long)ND * NDH);
  k_transpose_cvt<<<ew((long)NDH * ND), TPB, 0, stream>>>(W2, w2t, NDH, ND, (long)NDH * ND);

  k_state_init<<<ew((long)NB * NST * ND), TPB, 0, stream>>>(sinit, state);

  const long bt_elems = (long)NB * NT * ND;       // 1.31M
  const long rope_tot = (long)NB * NH * NT * 32;  // 327k
  const long vtr_tot  = (long)NB * NH * NT * 64;  // 655k

  // strides (elements)
  const long tm_sb = (long)NT * 512, tm_sh = 64;  const int tm_st = 512;  // token-major
  const long hm_sb = (long)NH * NT * 64, hm_sh = (long)NT * 64; const int hm_st = 64; // head-major

  for (int seg = 0; seg < NS / NL; ++seg) {
    k_build_toks<<<ew(bt_elems), TPB, 0, stream>>>(x, state, toks, seg);

    // stage-0 projections: [1280 x 1024] x [1024 x 512]
    dim3 gproj(512 / 128, NBT / 64, 1);
    gemm_bf16<64><<<gproj, 128, 0, stream>>>(toks, 0, ND, wq0t, 0, 1, ND,
                                             nullptr, q0, 0, 512, nullptr, 0, NBT, 512, ND);
    gemm_bf16<64><<<gproj, 128, 0, stream>>>(toks, 0, ND, wk0t, 0, 1, ND,
                                             nullptr, k0, 0, 512, nullptr, 0, NBT, 512, ND);
    gemm_bf16<64><<<gproj, 128, 0, stream>>>(toks, 0, ND, wv0t, 0, 1, ND,
                                             nullptr, v0, 0, 512, nullptr, 0, NBT, 512, ND);

    k_rope<<<ew(rope_tot), TPB, 0, stream>>>(q0, tm_sb, tm_sh, tm_st);
    k_rope<<<ew(rope_tot), TPB, 0, stream>>>(k0, tm_sb, tm_sh, tm_st);
    k_transpose_v<<<ew(vtr_tot), TPB, 0, stream>>>(v0, tm_sb, tm_sh, tm_st, v0t);

    dim3 gattn(NT / 16, NB * NH);
    attn_flash<<<gattn, 32, 0, stream>>>(q0, tm_sb, tm_sh, tm_st,
                                         k0, tm_sb, tm_sh, tm_st,
                                         v0t, o0, hm_sb, hm_sh, hm_st); // o0 head-major

    // stage-1 per-head projections: 16 batches of [640 x 64] x [64 x 64]
    dim3 ghead(1, NT / 64, NB * NH);
    gemm_bf16<32><<<ghead, 128, 0, stream>>>(o0, (long)NT * 64, 64, wq1t, 64 * 64, NH, 64,
                                             nullptr, q1, (long)NT * 64, 64, nullptr, 0, NT, 64, 64);
    gemm_bf16<32><<<ghead, 128, 0, stream>>>(o0, (long)NT * 64, 64, wk1t, 64 * 64, NH, 64,
                                             nullptr, k1, (long)NT * 64, 64, nullptr, 0, NT, 64, 64);
    gemm_bf16<32><<<ghead, 128, 0, stream>>>(o0, (long)NT * 64, 64, wv1t, 64 * 64, NH, 64,
                                             nullptr, v1, (long)NT * 64, 64, nullptr, 0, NT, 64, 64);

    k_rope<<<ew(rope_tot), TPB, 0, stream>>>(q1, hm_sb, hm_sh, hm_st);
    k_rope<<<ew(rope_tot), TPB, 0, stream>>>(k1, hm_sb, hm_sh, hm_st);
    k_transpose_v<<<ew(vtr_tot), TPB, 0, stream>>>(v1, hm_sb, hm_sh, hm_st, v1t);

    attn_flash<<<gattn, 32, 0, stream>>>(q1, hm_sb, hm_sh, hm_st,
                                         k1, hm_sb, hm_sh, hm_st,
                                         v1t, o1, tm_sb, tm_sh, tm_st); // o1 token-major

    // output projection: [1280 x 512] x [512 x 1024] + bo -> f32
    dim3 gout(ND / 128, NBT / 64, 1);
    gemm_bf16<64><<<gout, 128, 0, stream>>>(o1, 0, 512, wot, 0, 1, 512,
                                            outbuf, nullptr, 0, ND, bo, 0, NBT, ND, 512);

    k_scatter<<<ew(bt_elems), TPB, 0, stream>>>(outbuf, a_full, state, seg);
  }

  // h = LN(a + x) -> bf16
  k_layernorm<<<NBS, 256, 0, stream>>>(a_full, x, g_attn, b_attn, hbuf, nullptr);

  // m1 = gelu(h @ W1 + b1) -> bf16   [8192 x 1024] x [1024 x 4096]
  dim3 gm1(NDH / 128, NBS / 64, 1);
  gemm_bf16<64><<<gm1, 128, 0, stream>>>(hbuf, 0, ND, w1t, 0, 1, ND,
                                         nullptr, m1, 0, NDH, b1, 1, NBS, NDH, ND);
  // m2 = m1 @ W2 + b2 -> f32   [8192 x 4096] x [4096 x 1024]
  dim3 gm2(ND / 128, NBS / 64, 1);
  gemm_bf16<64><<<gm2, 128, 0, stream>>>(m1, 0, NDH, w2t, 0, 1, NDH,
                                         m2, nullptr, 0, ND, b2, 0, NBS, ND, NDH);

  // out = LN(m2 + x) -> f32
  k_layernorm<<<NBS, 256, 0, stream>>>(m2, x, g_mlp, b_mlp, nullptr, outp);
}
